// TuFormAttention_6073083756834
// MI455X (gfx1250) — matrix operand order repaired and checked by reading, not compiled, not run on hardware
//
#include <hip/hip_runtime.h>

// ---------------------------------------------------------------------------
// TuFormAttention fused kernels for MI455X (gfx1250, wave32, WMMA + TDM)
//
//   kernel 1:  qkv = X @ Wqkv^T + bqkv        (bf16 WMMA, f32 out; K also bf16)
//   kernel 2:  per (b,s): scores[l,p] = sum_r k[l,r]*(q[s,r]*Wc[p,r]*log2e)
//              out[s,p]  = sum_l softmax2_l(scores)[l,p] * v[l,p]
//   K tile staged global->LDS by the Tensor Data Mover (tensor_load_to_lds,
//   TENSORcnt), with hardware row padding matching the LDS layout.
//   (+bc is constant along the softmax axis l -> cancels exactly, dropped)
// ---------------------------------------------------------------------------

#define BATCH 2
#define SEQ   256
#define HID   256
#define QKV_N 768
#define PADR  272      // LDS row stride in bf16 elems: 512B row + 32B pad
#define LOG2E 1.4426950408889634f

typedef __attribute__((ext_vector_type(16))) __bf16 bf16x16;
typedef __attribute__((ext_vector_type(8)))  float  floatx8;
typedef __attribute__((ext_vector_type(4)))  unsigned int v4u;
typedef __attribute__((ext_vector_type(8)))  int    v8i_;
typedef __attribute__((ext_vector_type(4)))  int    v4i_;

union Frag {
    bf16x16        v;
    uint4          u[2];
    __bf16         b[16];
};

static __device__ __forceinline__ unsigned short f2bf_u16(float f) {
    // round-to-nearest-even f32 -> bf16 (raw u16, for plain stores)
    unsigned int u = __float_as_uint(f);
    u += 0x7FFFu + ((u >> 16) & 1u);
    return (unsigned short)(u >> 16);
}

// ---------------------------------------------------------------------------
// Kernel 1: qkv projection.  GEMM  M=512 (b*s), N=768, K=256.
// One wave per 16x16 output tile; 8 k-steps of v_wmma_f32_16x16x32_bf16.
// The K third (columns 256..511) is additionally written as bf16 for the TDM.
// ---------------------------------------------------------------------------
__global__ void __launch_bounds__(256) qkv_proj_kernel(
    const float* __restrict__ X,        // [512, 256]
    const float* __restrict__ W,        // [768, 256]
    const float* __restrict__ bias,     // [768]
    float* __restrict__ qkv,            // [512, 768] f32
    unsigned short* __restrict__ kbf)   // [512, 256] bf16 (K only)
{
    const int wid    = blockIdx.x * 8 + (threadIdx.x >> 5);  // 0..1535
    const int lane   = threadIdx.x & 31;
    const int lanelo = lane & 15;
    const int lanehi = lane >> 4;
    const int mt = wid / 48;            // 0..31
    const int nt = wid % 48;            // 0..47

    const int mrow = mt * 16 + lanelo;  // X row   (A: M = lane&15)
    const int ncol = nt * 16 + lanelo;  // W row   (B: N = lane&15)
    const float* __restrict__ xrow = X + (size_t)mrow * HID;
    const float* __restrict__ wrow = W + (size_t)ncol * HID;

    floatx8 c = {};
    for (int ks = 0; ks < 8; ++ks) {
        Frag a, b;
        // A 16-bit 16x32 layout: khi = lanehi*8; elems 0..7 -> K khi..khi+7,
        // elems 8..15 -> K 16+khi..16+khi+7
        const int ab = ks * 32 + lanehi * 8;
        #pragma unroll
        for (int e = 0; e < 8; ++e) a.b[e]     = (__bf16)xrow[ab + e];
        #pragma unroll
        for (int e = 0; e < 8; ++e) a.b[8 + e] = (__bf16)xrow[ab + 16 + e];
        // B 16-bit 32x16 layout: same K order as A (khi = lanehi*8, split 8+8)
        const int bb = ks * 32 + lanehi * 8;
        #pragma unroll
        for (int e = 0; e < 8; ++e) b.b[e]     = (__bf16)wrow[bb + e];
        #pragma unroll
        for (int e = 0; e < 8; ++e) b.b[8 + e] = (__bf16)wrow[bb + 16 + e];

        c = __builtin_amdgcn_wmma_f32_16x16x32_bf16(
                false, a.v, false, b.v, (short)0, c, false, false);
    }

    const float bv = bias[ncol];
    // D layout: N = lane&15, M = j + 8*lanehi
    #pragma unroll
    for (int j = 0; j < 8; ++j) {
        const int m = mt * 16 + j + lanehi * 8;
        const float r = c[j] + bv;
        qkv[(size_t)m * QKV_N + ncol] = r;
        if (nt >= 16 && nt < 32)                        // K third -> bf16 copy
            kbf[(size_t)m * HID + (ncol - 256)] = f2bf_u16(r);
    }
}

// ---------------------------------------------------------------------------
// TDM: DMA one 256x256 bf16 tile (512B rows) global -> LDS, inserting 32B of
// pad after every 512B so LDS rows land at the PADR=272-element stride.
// D# built per CDNA5 ISA ch.8: group0 {count,lds,global,type}, group1
// {data_size=2B, pad 8DW every 128DW, dims/tile 256x256, dim0_stride=256}.
// ---------------------------------------------------------------------------
static __device__ __forceinline__ void tdm_load_k_tile(
    const unsigned short* __restrict__ gsrc, unsigned lds_off)
{
    const unsigned long long ga = (unsigned long long)(uintptr_t)gsrc;
    v4u g0;
    g0[0] = 1u;                                        // count=1 (valid user D#)
    g0[1] = lds_off;                                   // lds_addr (bytes)
    g0[2] = (unsigned)(ga & 0xFFFFFFFFu);              // global_addr[31:0]
    g0[3] = (unsigned)((ga >> 32) & 0x1FFFFFFu)        // global_addr[56:32]
          | (2u << 30);                                // type=2 ("image")
    v8i_ g1;
    g1[0] = (1 << 16)                                  // data_size = 2 bytes
          | (1 << 20)                                  // pad_enable
          | (6 << 22)                                  // pad_interval: 128 DW
          | (7 << 25);                                 // pad_amount: 8 DW
    g1[1] = (int)(256u << 16);                         // tensor_dim0 = 256
    g1[2] = (int)(256u << 16);                         // tensor_dim1 = 256
    g1[3] = (int)(256u << 16);                         // tile_dim0 = 256
    g1[4] = 256;                                       // tile_dim1 = 256
    g1[5] = 256;                                       // tensor_dim0_stride=256
    g1[6] = 0;                                         // dim1_stride[15:0]<<16
    g1[7] = 1;                                         // dim1_stride = 65536
    const v4i_ gz = {0, 0, 0, 0};
#if __clang_major__ >= 23
    const v8i_ gz8 = {0, 0, 0, 0, 0, 0, 0, 0};
    __builtin_amdgcn_tensor_load_to_lds(g0, g1, gz, gz, gz8, 0);
#else
    __builtin_amdgcn_tensor_load_to_lds(g0, g1, gz, gz, 0);
#endif
}

// ---------------------------------------------------------------------------
// Kernel 2: fused scores-GEMM + online softmax(l) + V reduction.
// One workgroup per (b,s); 8 waves; wave w owns output columns [32w, 32w+32).
// LDS: K[b] bf16 256 x PADR (136 KB) staged by the TDM. B operands
// (log2e * q * Wc, bf16) live in registers for the whole kernel.
// ---------------------------------------------------------------------------
__global__ void __launch_bounds__(256) tuform_attn_kernel(
    const float* __restrict__ qkv,            // [512, 768] f32
    const unsigned short* __restrict__ kbf,   // [512, 256] bf16
    const float* __restrict__ Wc,             // [256, 256]
    float* __restrict__ out)                  // [512, 256]
{
    extern __shared__ unsigned short kls[];   // [256][PADR] bf16

    const int bs = blockIdx.x;                // 0..511
    const int b  = bs >> 8;
    const int s  = bs & 255;
    const int tid = threadIdx.x;

    const float* __restrict__ qrow  = qkv + (size_t)(b * SEQ + s) * QKV_N;   // q[s,r]
    const float* __restrict__ vbase = qkv + (size_t)(b * SEQ) * QKV_N + 512; // v[l,p]

    const int lane   = tid & 31;
    const int wave   = tid >> 5;
    const int lanelo = lane & 15;
    const int lanehi = lane >> 4;

    // ---- stage K[b] into LDS via the Tensor Data Mover (one wave issues)
    if (wave == 0) {
        tdm_load_k_tile(kbf + (size_t)b * SEQ * HID, (unsigned)(uintptr_t)kls);
        __builtin_amdgcn_s_wait_tensorcnt(0);
    }

    // ---- preload B fragments (log2e * q * Wc), 2 p-tiles x 8 k-steps
    Frag bfr[2][8];
    #pragma unroll
    for (int pt = 0; pt < 2; ++pt) {
        const int p = wave * 32 + pt * 16 + lanelo;
        const float* __restrict__ wrow = Wc + (size_t)p * HID;
        for (int ks = 0; ks < 8; ++ks) {
            const int r0 = ks * 32 + lanehi * 8;
            #pragma unroll
            for (int e = 0; e < 16; ++e)
                bfr[pt][ks].b[e] = (__bf16)(LOG2E * qrow[r0 + e + (e >= 8 ? 8 : 0)] * wrow[r0 + e + (e >= 8 ? 8 : 0)]);
        }
    }

    __syncthreads();   // TDM data visible to all waves

    const int p0 = wave * 32 + lanelo;
    const int p1 = wave * 32 + 16 + lanelo;

    float m0 = -INFINITY, d0 = 0.f, ac0 = 0.f;
    float m1 = -INFINITY, d1 = 0.f, ac1 = 0.f;

    for (int lt = 0; lt < 16; ++lt) {
        floatx8 c0 = {}, c1 = {};
        const int arow = (lt * 16 + lanelo) * PADR + lanehi * 8;
        for (int ks = 0; ks < 8; ++ks) {
            Frag a;
            a.u[0] = *(const uint4*)(&kls[arow + ks * 32]);      // K ..+7
            a.u[1] = *(const uint4*)(&kls[arow + ks * 32 + 16]); // K +16..+23
            c0 = __builtin_amdgcn_wmma_f32_16x16x32_bf16(
                     false, a.v, false, bfr[0][ks].v, (short)0, c0, false, false);
            c1 = __builtin_amdgcn_wmma_f32_16x16x32_bf16(
                     false, a.v, false, bfr[1][ks].v, (short)0, c1, false, false);
        }

        // ---- online base-2 softmax over l for both 16x16 tiles
        const int lb = lt * 16 + lanehi * 8;   // this lane's 8 l-rows
        #pragma unroll
        for (int t = 0; t < 2; ++t) {
            const floatx8 c = t ? c1 : c0;
            const int p = t ? p1 : p0;
            float& m  = t ? m1 : m0;
            float& d  = t ? d1 : d0;
            float& ac = t ? ac1 : ac0;

            float tmax = c[0];
            #pragma unroll
            for (int j = 1; j < 8; ++j) tmax = fmaxf(tmax, c[j]);
            tmax = fmaxf(tmax, __shfl_xor(tmax, 16, 32));   // join lane halves
            const float mn = fmaxf(m, tmax);
            const float sc = exp2f(m - mn);

            float ps = 0.f, pa = 0.f;
            const float* __restrict__ vcol = vbase + p;
            #pragma unroll
            for (int j = 0; j < 8; ++j) {
                const float e = exp2f(c[j] - mn);
                ps += e;
                pa += e * vcol[(size_t)(lb + j) * QKV_N];
            }
            ps += __shfl_xor(ps, 16, 32);
            pa += __shfl_xor(pa, 16, 32);

            d  = d * sc + ps;
            ac = ac * sc + pa;
            m  = mn;
        }
    }

    if (lanehi == 0) {
        float* __restrict__ orow = out + (size_t)(b * SEQ + s) * HID;
        orow[p0] = ac0 / d0;
        orow[p1] = ac1 / d1;
    }
}

// ---------------------------------------------------------------------------
extern "C" void kernel_launch(void* const* d_in, const int* in_sizes, int n_in,
                              void* d_out, int out_size, void* d_ws, size_t ws_size,
                              hipStream_t stream) {
    const float* hidden = (const float*)d_in[0];   // [2,256,256]
    const float* Wqkv   = (const float*)d_in[1];   // [768,256]
    const float* bqkv   = (const float*)d_in[2];   // [768]
    const float* Wc     = (const float*)d_in[3];   // [256,256]
    // d_in[4] = bc: constant along the softmax axis -> cancels, unused.
    float* out = (float*)d_out;                    // [2,256,256]

    float*          qkvws = (float*)d_ws;                          // 1.5 MB
    unsigned short* kbfws = (unsigned short*)((char*)d_ws +
                            (size_t)BATCH * SEQ * QKV_N * sizeof(float)); // +256 KB

    (void)in_sizes; (void)n_in; (void)out_size; (void)ws_size;

    const size_t lds_bytes = (size_t)SEQ * PADR * sizeof(unsigned short); // 139264
    (void)hipFuncSetAttribute((const void*)tuform_attn_kernel,
                              hipFuncAttributeMaxDynamicSharedMemorySize,
                              (int)lds_bytes);

    qkv_proj_kernel<<<192, 256, 0, stream>>>(hidden, Wqkv, bqkv, qkvws, kbfws);
    tuform_attn_kernel<<<BATCH * SEQ, 256, lds_bytes, stream>>>(qkvws, kbfws, Wc, out);
}
